// DiffusionStepModel_89481348644994
// MI455X (gfx1250) — matrix operand order repaired and checked by reading, not compiled, not run on hardware
//
#include <hip/hip_runtime.h>
#include <hip/hip_bf16.h>

// ---------------------------------------------------------------------------
// DiffusionStepModel forward for MI455X (gfx1250, wave32, WMMA).
// f16 operand storage + f32 WMMA accumulation; LayerNorm fused into the GEMM
// epilogue; message GEMM fused with the sender-gather and receiver-scatter
// (atomic f32 segment-sum). Entire MP working set (~95 MB) is L2-resident.
// ---------------------------------------------------------------------------

#define NNODES 30000
#define NEDGES 480000

typedef __attribute__((ext_vector_type(16))) _Float16 v16h;
typedef __attribute__((ext_vector_type(8)))  float    v8f;

union ABPack { v16h v; uint4 u[2]; };

// A-fragment (16x32 f16): lo lanes hold K {kc*32+0..7, +16..23}, hi lanes +8.
__device__ __forceinline__ v16h load_a_frag(const _Float16* __restrict__ rowp,
                                            int kc, int hi) {
  const _Float16* p = rowp + kc * 32 + (hi ? 8 : 0);
  ABPack t;
  t.u[0] = *(const uint4*)(p);
  t.u[1] = *(const uint4*)(p + 16);
  return t.v;
}

// B-fragment (32x16 f16): column n = lane&15; lo lanes K {kc*32+0..7, +16..23},
// hi lanes +8. Row-major W (out=n, in=k) IS B = W^T.
__device__ __forceinline__ v16h load_b_frag(const _Float16* __restrict__ W,
                                            int K, int n, int kc, int hi) {
  const _Float16* p = W + (size_t)n * K + kc * 32 + (hi ? 8 : 0);
  ABPack t;
  t.u[0] = *(const uint4*)(p);
  t.u[1] = *(const uint4*)(p + 16);
  return t.v;
}

// ---------------------------------------------------------------------------
// Fused GEMM (M x K @ K x 64) + [bias] + [ReLU] + [residual add] + LayerNorm.
// One wave = 16 output rows x 64 cols (4 C-fragments). Per-row LN via
// shfl_xor reductions inside each 16-lane half (wave32; masks 1/2/4/8 stay
// inside the half), plus the 4 fragments held by the same lanes.
// ---------------------------------------------------------------------------
template <bool RELU, bool HAS_BIAS, bool HAS_ADD>
__global__ void gemm_ln_kernel(const _Float16* __restrict__ A,
                               const _Float16* __restrict__ W,
                               const float* __restrict__ bias,
                               const _Float16* __restrict__ addend,
                               const float* __restrict__ g,
                               const float* __restrict__ be,
                               _Float16* __restrict__ out, int M, int K) {
  const int wave = threadIdx.x >> 5;
  const int lane = threadIdx.x & 31;
  const int tileM = blockIdx.x * 8 + wave;
  const int row0 = tileM * 16;
  if (row0 >= M) return;                 // wave-uniform (EXEC stays all-1s)
  const int ln = lane & 15;
  const int hi = lane >> 4;

  v8f c0 = {}, c1 = {}, c2 = {}, c3 = {};
  const _Float16* arow = A + (size_t)(row0 + ln) * K;
  const int nK = K >> 5;
  for (int kc = 0; kc < nK; ++kc) {
    v16h a = load_a_frag(arow, kc, hi);
    v16h b0 = load_b_frag(W, K, 0 * 16 + ln, kc, hi);
    v16h b1 = load_b_frag(W, K, 1 * 16 + ln, kc, hi);
    v16h b2 = load_b_frag(W, K, 2 * 16 + ln, kc, hi);
    v16h b3 = load_b_frag(W, K, 3 * 16 + ln, kc, hi);
    c0 = __builtin_amdgcn_wmma_f32_16x16x32_f16(false, a, false, b0, (short)0, c0, false, false);
    c1 = __builtin_amdgcn_wmma_f32_16x16x32_f16(false, a, false, b1, (short)0, c1, false, false);
    c2 = __builtin_amdgcn_wmma_f32_16x16x32_f16(false, a, false, b2, (short)0, c2, false, false);
    c3 = __builtin_amdgcn_wmma_f32_16x16x32_f16(false, a, false, b3, (short)0, c3, false, false);
  }
  v8f c[4] = {c0, c1, c2, c3};

  // epilogue: bias / relu / residual, per element (M = r + 8*hi, N = t*16+ln)
  float bn[4];
#pragma unroll
  for (int t = 0; t < 4; ++t) bn[t] = HAS_BIAS ? bias[t * 16 + ln] : 0.f;
#pragma unroll
  for (int t = 0; t < 4; ++t) {
#pragma unroll
    for (int r = 0; r < 8; ++r) {
      float x = c[t][r] + bn[t];
      if (RELU) x = fmaxf(x, 0.f);
      if (HAS_ADD)
        x += (float)addend[(size_t)(row0 + r + 8 * hi) * 64 + t * 16 + ln];
      c[t][r] = x;
    }
  }
  // per-row LayerNorm over 64 cols
#pragma unroll
  for (int r = 0; r < 8; ++r) {
    float s = c[0][r] + c[1][r] + c[2][r] + c[3][r];
    float q = c[0][r] * c[0][r] + c[1][r] * c[1][r] + c[2][r] * c[2][r] + c[3][r] * c[3][r];
#pragma unroll
    for (int m = 1; m < 16; m <<= 1) {
      s += __shfl_xor(s, m, 32);
      q += __shfl_xor(q, m, 32);
    }
    const float mean = s * (1.f / 64.f);
    const float var = q * (1.f / 64.f) - mean * mean;
    const float rstd = rsqrtf(var + 1e-5f);
    const size_t orow = (size_t)(row0 + r + 8 * hi) * 64;
#pragma unroll
    for (int t = 0; t < 4; ++t) {
      const int n = t * 16 + ln;
      out[orow + n] = (_Float16)((c[t][r] - mean) * rstd * g[n] + be[n]);
    }
  }
}

// ---------------------------------------------------------------------------
// Fused message step: gather nodes[sender] (K chunks 0-1) + edge feats
// (K chunks 2-3), WMMA against W_msg (64x128), atomic scatter-add into
// agg[receiver].  One wave = 16 edges x 64 message dims.
// ---------------------------------------------------------------------------
__global__ void msg_scatter_kernel(const _Float16* __restrict__ nodesH,
                                   const _Float16* __restrict__ edgesH,
                                   const _Float16* __restrict__ Wmsg,
                                   const int* __restrict__ eidx,  // [2][E]
                                   float* __restrict__ agg, int E) {
  const int wave = threadIdx.x >> 5;
  const int lane = threadIdx.x & 31;
  const int tile = blockIdx.x * 8 + wave;
  const int e0 = tile * 16;
  if (e0 >= E) return;
  const int ln = lane & 15;
  const int hi = lane >> 4;
  const int myE = e0 + ln;
  const int sender = eidx[myE];

  const _Float16* arow0 = nodesH + (size_t)sender * 64;
  const _Float16* arow1 = edgesH + (size_t)myE * 64;

  v8f c[4] = {};
#pragma unroll
  for (int kc = 0; kc < 4; ++kc) {
    const _Float16* base = (kc < 2) ? (arow0 + kc * 32) : (arow1 + (kc - 2) * 32);
    const _Float16* p = base + (hi ? 8 : 0);
    ABPack t;
    t.u[0] = *(const uint4*)(p);
    t.u[1] = *(const uint4*)(p + 16);
    v16h a = t.v;
#pragma unroll
    for (int tn = 0; tn < 4; ++tn) {
      v16h b = load_b_frag(Wmsg, 128, tn * 16 + ln, kc, hi);
      c[tn] = __builtin_amdgcn_wmma_f32_16x16x32_f16(false, a, false, b, (short)0, c[tn], false, false);
    }
  }
  // segment_sum: atomic f32 scatter by receiver
#pragma unroll
  for (int r = 0; r < 8; ++r) {
    const int e = e0 + r + 8 * hi;
    const int recv = eidx[E + e];
    float* dst = agg + (size_t)recv * 64;
#pragma unroll
    for (int tn = 0; tn < 4; ++tn) atomicAdd(dst + tn * 16 + ln, c[tn][r]);
  }
}

// ---------------------------------------------------------------------------
// Small helper kernels
// ---------------------------------------------------------------------------
__global__ void cvt_f32_f16(const float* __restrict__ s, _Float16* __restrict__ d, int n) {
  int i = blockIdx.x * blockDim.x + threadIdx.x;
  if (i < n) d[i] = (_Float16)s[i];
}

// node_enc layer-0 weight: (64 x 39) f32 -> (64 x 64) f16, zero-padded K.
__global__ void cvt_pad_w1(const float* __restrict__ s, _Float16* __restrict__ d) {
  int i = blockIdx.x * blockDim.x + threadIdx.x;
  if (i >= 64 * 64) return;
  int n = i >> 6, k = i & 63;
  d[i] = (k < 39) ? (_Float16)s[n * 39 + k] : (_Float16)0.f;
}

__global__ void zero_f32(float* __restrict__ p, int n) {
  int i = blockIdx.x * blockDim.x + threadIdx.x;
  if (i < n) p[i] = 0.f;
}

// x_in = [one_hot(X_t,2) | time_pe(32) | rand(5) | zero-pad to 64] as f16
__global__ void build_xin_kernel(const int* __restrict__ Xt, const int* __restrict__ ts,
                                 const float* __restrict__ rnd, _Float16* __restrict__ out,
                                 int N) {
  int idx = blockIdx.x * blockDim.x + threadIdx.x;
  if (idx >= N * 64) return;
  int row = idx >> 6, col = idx & 63;
  float v = 0.f;
  if (col < 2) {
    v = (Xt[row] == col) ? 1.f : 0.f;
  } else if (col < 34) {
    int j = col - 2;
    int i = j >> 1;
    float div = expf(-(float)(2 * i) * 0.2878231366242557f);  // ln(1e4)/32
    float ang = (float)ts[0] * div;
    v = (j & 1) ? cosf(ang) : sinf(ang);
  } else if (col < 39) {
    v = rnd[row * 5 + (col - 34)];
  }
  out[idx] = (_Float16)v;
}

// cat = [nodes(f16) | f16(agg f32)]  (N x 128)
__global__ void cat_kernel(const _Float16* __restrict__ nodesH, const float* __restrict__ agg,
                           _Float16* __restrict__ cat, int N) {
  int idx = blockIdx.x * blockDim.x + threadIdx.x;
  if (idx >= N * 128) return;
  int row = idx >> 7, col = idx & 127;
  cat[idx] = (col < 64) ? nodesH[(size_t)row * 64 + col]
                        : (_Float16)agg[(size_t)row * 64 + (col - 64)];
}

// edge encoder layer 0: K=2 -> 64, ReLU + LN per edge (one thread per edge)
__global__ void edge_enc1_kernel(const float* __restrict__ ea, const float* __restrict__ W,
                                 const float* __restrict__ b, const float* __restrict__ g,
                                 const float* __restrict__ be, _Float16* __restrict__ out,
                                 int E) {
  int e = blockIdx.x * blockDim.x + threadIdx.x;
  if (e >= E) return;
  float a0 = ea[2 * e], a1 = ea[2 * e + 1];
  float v[64];
  float s = 0.f, q = 0.f;
#pragma unroll
  for (int n = 0; n < 64; ++n) {
    float x = fmaxf(W[2 * n] * a0 + W[2 * n + 1] * a1 + b[n], 0.f);
    v[n] = x;
    s += x;
    q += x * x;
  }
  float mean = s * (1.f / 64.f);
  float var = q * (1.f / 64.f) - mean * mean;
  float rstd = rsqrtf(var + 1e-5f);
  _Float16* o = out + (size_t)e * 64;
#pragma unroll
  for (int n = 0; n < 64; ++n) o[n] = (_Float16)((v[n] - mean) * rstd * g[n] + be[n]);
}

// prob out head: logits (64 -> 2) + log_softmax, one thread per node
__global__ void prob_out_kernel(const _Float16* __restrict__ ph, const float* __restrict__ Wp,
                                const float* __restrict__ bp, float* __restrict__ out, int N) {
  int i = blockIdx.x * blockDim.x + threadIdx.x;
  if (i >= N) return;
  const _Float16* r = ph + (size_t)i * 64;
  float l0 = bp[0], l1 = bp[1];
#pragma unroll
  for (int k = 0; k < 64; ++k) {
    float x = (float)r[k];
    l0 += x * Wp[k];
    l1 += x * Wp[64 + k];
  }
  float m = fmaxf(l0, l1);
  float lse = m + logf(expf(l0 - m) + expf(l1 - m));
  out[2 * i] = l0 - lse;
  out[2 * i + 1] = l1 - lse;
}

// column sums of nodes_dec for mean pooling (atomic into pooled[64])
__global__ void colsum_kernel(const _Float16* __restrict__ x, float* __restrict__ pooled, int M) {
  __shared__ float sm[256];
  int col = threadIdx.x & 63;
  int rg = threadIdx.x >> 6;  // 0..3
  float s = 0.f;
  int base = blockIdx.x * 256;
  for (int i = 0; i < 64; ++i) {
    int r = base + rg + 4 * i;
    if (r < M) s += (float)x[(size_t)r * 64 + col];
  }
  sm[threadIdx.x] = s;
  __syncthreads();
  if (rg == 0) {
    float t = sm[col] + sm[col + 64] + sm[col + 128] + sm[col + 192];
    atomicAdd(pooled + col, t);
  }
}

// value head: mean -> Linear+ReLU+LN (64) -> Linear(64->1).  One block, 64 thr.
__global__ void value_kernel(const float* __restrict__ pooled, const float* __restrict__ Wh,
                             const float* __restrict__ bh, const float* __restrict__ g,
                             const float* __restrict__ be, const float* __restrict__ Wv,
                             const float* __restrict__ bv, float* __restrict__ out, int M) {
  __shared__ float sh[64];
  int n = threadIdx.x;
  float acc = bh[n];
  float invM = 1.f / (float)M;
  for (int k = 0; k < 64; ++k) acc += Wh[n * 64 + k] * (pooled[k] * invM);
  acc = fmaxf(acc, 0.f);
  sh[n] = acc;
  __syncthreads();
  float s = 0.f, q = 0.f;
  for (int k = 0; k < 64; ++k) {
    float x = sh[k];
    s += x;
    q += x * x;
  }
  float mean = s * (1.f / 64.f);
  float var = q * (1.f / 64.f) - mean * mean;
  float rstd = rsqrtf(var + 1e-5f);
  float y = (acc - mean) * rstd * g[n] + be[n];
  __syncthreads();
  sh[n] = y * Wv[n];
  __syncthreads();
  if (n == 0) {
    float v = bv[0];
    for (int k = 0; k < 64; ++k) v += sh[k];
    out[0] = v;
  }
}

// ---------------------------------------------------------------------------
// Host driver.  Param flattening assumed jax-pytree (sorted dict keys):
//   [X_t, timestep, rand_nodes, edge_index, edge_attr,
//    edge_enc(8), layers[0..4]{W_msg,W_node,ln_g,ln_b,node_mlp(8)}=12 each,
//    node_dec(8), node_enc(8), prob{hidden(4),outW,outB},
//    value{hidden(4),outW,outB}]  -> 101 entries.
// ---------------------------------------------------------------------------
enum {
  I_XT = 0, I_TS, I_RAND, I_EIDX, I_EATTR,
  EE = 5,            // W1,b1,g1,be1, W2,b2,g2,be2
  LYR = EE + 8,      // 5 x 12
  ND = LYR + 60,     // W1,b1,g1,be1, W2,b2,g2,be2
  NE = ND + 8,
  PR = NE + 8,       // hW,hb,hg,hbe, oW,ob
  VA = PR + 6,
};
#define LWI(l, j) (LYR + (l) * 12 + (j))
// j: 0 W_msg, 1 W_node, 2 ln_g, 3 ln_b, 4 nmW1,5 nmB1,6 nmG1,7 nmBE1, 8 nmW2,9 nmB2,10 nmG2,11 nmBE2

extern "C" void kernel_launch(void* const* d_in, const int* in_sizes, int n_in,
                              void* d_out, int out_size, void* d_ws, size_t ws_size,
                              hipStream_t stream) {
  const int N = NNODES, E = NEDGES;
  const int* Xt = (const int*)d_in[I_XT];
  const int* ts = (const int*)d_in[I_TS];
  const float* rnd = (const float*)d_in[I_RAND];
  const int* eidx = (const int*)d_in[I_EIDX];
  const float* ea = (const float*)d_in[I_EATTR];
  auto F = [&](int i) { return (const float*)d_in[i]; };

  // ---- workspace carve-up (all offsets 256B aligned) ----
  uintptr_t wp = (uintptr_t)d_ws;
  auto alloc = [&](size_t bytes) -> void* {
    void* p = (void*)wp;
    wp += (bytes + 255) & ~(size_t)255;
    return p;
  };
  _Float16* h1 = (_Float16*)alloc((size_t)N * 64 * 2);
  _Float16* h2 = (_Float16*)alloc((size_t)N * 64 * 2);
  _Float16* nodesA = (_Float16*)alloc((size_t)N * 64 * 2);
  _Float16* nodesB = (_Float16*)alloc((size_t)N * 64 * 2);
  _Float16* nodeout = (_Float16*)alloc((size_t)N * 64 * 2);
  _Float16* cat_h = (_Float16*)alloc((size_t)N * 128 * 2);
  float* agg = (float*)alloc((size_t)N * 64 * 4);
  _Float16* edges = (_Float16*)alloc((size_t)E * 64 * 2);
  float* pooled = (float*)alloc(256);
  _Float16* neW1p = (_Float16*)alloc(4096 * 2);
  _Float16* neW2 = (_Float16*)alloc(4096 * 2);
  _Float16* eeW2 = (_Float16*)alloc(4096 * 2);
  _Float16 *Wmsg[5], *Wnode[5], *nmW1[5], *nmW2[5];
  for (int l = 0; l < 5; ++l) {
    Wmsg[l] = (_Float16*)alloc(8192 * 2);
    Wnode[l] = (_Float16*)alloc(4096 * 2);
    nmW1[l] = (_Float16*)alloc(8192 * 2);
    nmW2[l] = (_Float16*)alloc(4096 * 2);
  }
  _Float16* ndW1 = (_Float16*)alloc(4096 * 2);
  _Float16* ndW2 = (_Float16*)alloc(4096 * 2);
  _Float16* prW = (_Float16*)alloc(4096 * 2);

  // ---- weight conversion to f16 ----
  cvt_pad_w1<<<16, 256, 0, stream>>>(F(NE + 0), neW1p);
  cvt_f32_f16<<<16, 256, 0, stream>>>(F(NE + 4), neW2, 4096);
  cvt_f32_f16<<<16, 256, 0, stream>>>(F(EE + 4), eeW2, 4096);
  for (int l = 0; l < 5; ++l) {
    cvt_f32_f16<<<32, 256, 0, stream>>>(F(LWI(l, 0)), Wmsg[l], 8192);
    cvt_f32_f16<<<16, 256, 0, stream>>>(F(LWI(l, 1)), Wnode[l], 4096);
    cvt_f32_f16<<<32, 256, 0, stream>>>(F(LWI(l, 4)), nmW1[l], 8192);
    cvt_f32_f16<<<16, 256, 0, stream>>>(F(LWI(l, 8)), nmW2[l], 4096);
  }
  cvt_f32_f16<<<16, 256, 0, stream>>>(F(ND + 0), ndW1, 4096);
  cvt_f32_f16<<<16, 256, 0, stream>>>(F(ND + 4), ndW2, 4096);
  cvt_f32_f16<<<16, 256, 0, stream>>>(F(PR + 0), prW, 4096);

  auto gblk = [](int M) { return (M / 16 + 7) / 8; };  // M is a multiple of 16

  // ---- encode ----
  build_xin_kernel<<<(N * 64 + 255) / 256, 256, 0, stream>>>(Xt, ts, rnd, h1, N);
  gemm_ln_kernel<true, true, false><<<gblk(N), 256, 0, stream>>>(
      h1, neW1p, F(NE + 1), nullptr, F(NE + 2), F(NE + 3), h2, N, 64);
  gemm_ln_kernel<true, true, false><<<gblk(N), 256, 0, stream>>>(
      h2, neW2, F(NE + 5), nullptr, F(NE + 6), F(NE + 7), nodesA, N, 64);

  edge_enc1_kernel<<<(E + 255) / 256, 256, 0, stream>>>(
      ea, F(EE + 0), F(EE + 1), F(EE + 2), F(EE + 3), edges, E);
  // layer 2 in place (each wave reads only the 16 rows it writes)
  gemm_ln_kernel<true, true, false><<<gblk(E), 256, 0, stream>>>(
      edges, eeW2, F(EE + 5), nullptr, F(EE + 6), F(EE + 7), edges, E, 64);

  // ---- message passing ----
  _Float16* cur = nodesA;
  _Float16* nxt = nodesB;
  for (int l = 0; l < 5; ++l) {
    zero_f32<<<(N * 64 + 255) / 256, 256, 0, stream>>>(agg, N * 64);
    msg_scatter_kernel<<<gblk(E), 256, 0, stream>>>(cur, edges, Wmsg[l], eidx, agg, E);
    cat_kernel<<<(N * 128 + 255) / 256, 256, 0, stream>>>(cur, agg, cat_h, N);
    gemm_ln_kernel<true, true, false><<<gblk(N), 256, 0, stream>>>(
        cat_h, nmW1[l], F(LWI(l, 5)), nullptr, F(LWI(l, 6)), F(LWI(l, 7)), h1, N, 128);
    gemm_ln_kernel<true, true, false><<<gblk(N), 256, 0, stream>>>(
        h1, nmW2[l], F(LWI(l, 9)), nullptr, F(LWI(l, 10)), F(LWI(l, 11)), nodeout, N, 64);
    gemm_ln_kernel<false, false, true><<<gblk(N), 256, 0, stream>>>(
        cur, Wnode[l], nullptr, nodeout, F(LWI(l, 2)), F(LWI(l, 3)), nxt, N, 64);
    _Float16* t = cur; cur = nxt; nxt = t;
  }

  // ---- decode ----
  gemm_ln_kernel<true, true, false><<<gblk(N), 256, 0, stream>>>(
      cur, ndW1, F(ND + 1), nullptr, F(ND + 2), F(ND + 3), h1, N, 64);
  gemm_ln_kernel<true, true, false><<<gblk(N), 256, 0, stream>>>(
      h1, ndW2, F(ND + 5), nullptr, F(ND + 6), F(ND + 7), h2, N, 64);  // nodes_dec

  // ---- prob head ----
  gemm_ln_kernel<true, true, false><<<gblk(N), 256, 0, stream>>>(
      h2, prW, F(PR + 1), nullptr, F(PR + 2), F(PR + 3), h1, N, 64);
  prob_out_kernel<<<(N + 255) / 256, 256, 0, stream>>>(h1, F(PR + 4), F(PR + 5),
                                                       (float*)d_out, N);

  // ---- value head ----
  zero_f32<<<1, 64, 0, stream>>>(pooled, 64);
  colsum_kernel<<<(N + 255) / 256, 256, 0, stream>>>(h2, pooled, N);
  value_kernel<<<1, 64, 0, stream>>>(pooled, F(VA + 0), F(VA + 1), F(VA + 2), F(VA + 3),
                                     F(VA + 4), F(VA + 5), (float*)d_out + 2 * N, N);

  (void)in_sizes; (void)n_in; (void)out_size; (void)ws_size;
}
